// SpatialConvolutionMap_25649544691935
// MI455X (gfx1250) — matrix operand order repaired and checked by reading, not compiled, not run on hardware
//
#include <hip/hip_runtime.h>

// ---------------------------------------------------------------------------
// SpatialConvolutionMap on MI455X (gfx1250, wave32, WMMA).
//
// Reformulated as a dense 64->64 3x3 conv with effective weights
//   W_eff[o][c][kh][kw] = sum_{t: conn_in[o*16+t]==c} weight[o*16+t][kh][kw]
// then implicit GEMM:  C(16 ch x 16 px) += A(16 ch x 32 k) * B(32 k x 16 px)
// with split-bf16 (hi+lo) inputs and f32 accumulation:
//   acc += Ah*Bh + Ah*Bl + Al*Bh     (error ~2^-16 relative)
// ---------------------------------------------------------------------------

typedef __bf16 v16bf __attribute__((ext_vector_type(16)));
typedef float  v8f   __attribute__((ext_vector_type(8)));

constexpr int N_IN  = 64;
constexpr int N_OUT = 64;
constexpr int N_TO  = 16;
constexpr int KTAPS = 9;                 // 3x3
constexpr int KDIM  = N_IN * KTAPS;      // 576 = 18 * 32
constexpr int NB    = 4;
constexpr int H     = 192;
constexpr int W     = 192;
constexpr int HO    = 190;
constexpr int WO    = 190;

__device__ __forceinline__ unsigned short bf16_rne(float x) {
    unsigned int u = __float_as_uint(x);
    u += 0x7FFFu + ((u >> 16) & 1u);
    return (unsigned short)(u >> 16);
}
__device__ __forceinline__ float bf16_as_f32(unsigned short h) {
    return __uint_as_float(((unsigned int)h) << 16);
}

// ---------------------------------------------------------------------------
// Kernel 1: build effective dense weights, split into bf16 hi/lo planes.
// Layout: Wh[o*KDIM + c*9 + (kh*3+kw)], same for Wl.  Deterministic (no
// float atomics): each k-entry scans its channel's 16 taps in fixed order.
// ---------------------------------------------------------------------------
__global__ void prep_weights(const float* __restrict__ weight,
                             const int*   __restrict__ conn_in,
                             const int*   __restrict__ conn_out,
                             unsigned short* __restrict__ Wh,
                             unsigned short* __restrict__ Wl) {
    const int o = blockIdx.x;            // one block per output channel
    for (int k = threadIdx.x; k < KDIM; k += blockDim.x) {
        const int c   = k / KTAPS;
        const int tap = k - c * KTAPS;
        float s = 0.0f;
        const int t0 = o * N_TO;
        for (int t = t0; t < t0 + N_TO; ++t) {
            if (conn_out[t] == o && conn_in[t] == c)
                s += weight[t * KTAPS + tap];
        }
        const unsigned short h = bf16_rne(s);
        const unsigned short l = bf16_rne(s - bf16_as_f32(h));
        Wh[o * KDIM + k] = h;
        Wl[o * KDIM + k] = l;
    }
}

// ---------------------------------------------------------------------------
// Kernel 2: implicit-GEMM conv.  One 128-thread block (4 wave32s) computes a
// tile of 16 output pixels (one row, 16 consecutive columns) x all 64 output
// channels.  Wave w handles channel tile w (16 channels) with one 16x16 f32
// accumulator; all waves share the im2col patch matrix staged in LDS.
// ---------------------------------------------------------------------------
__global__ __launch_bounds__(128)
void conv_wmma(const float* __restrict__ x,
               const float* __restrict__ bias,
               const unsigned short* __restrict__ Wh,
               const unsigned short* __restrict__ Wl,
               float* __restrict__ out) {
    // Patch matrix: P[pixel 0..15][k 0..575], bf16 hi/lo planes (36 KB LDS).
    __shared__ __align__(16) unsigned short Ph[16 * KDIM];
    __shared__ __align__(16) unsigned short Pl[16 * KDIM];

    const int j0  = blockIdx.x * 16;     // output column tile
    const int i   = blockIdx.y;          // output row
    const int b   = blockIdx.z;          // batch
    const int tid = threadIdx.x;

    // ---- im2col expansion into LDS (hi/lo split done once per element) ----
    for (int e = tid; e < 16 * KDIM; e += 128) {
        const int p   = e / KDIM;
        const int k   = e - p * KDIM;
        const int c   = k / KTAPS;
        const int tap = k - c * KTAPS;
        const int kh  = tap / 3;
        const int kw  = tap - kh * 3;
        int jx = j0 + p + kw;            // <= 193 on the last tile
        if (jx > W - 1) jx = W - 1;      // clamped lanes are never stored
        const float v = x[((b * N_IN + c) * H + (i + kh)) * W + jx];
        const unsigned short h = bf16_rne(v);
        Ph[e] = h;
        Pl[e] = bf16_rne(v - bf16_as_f32(h));
    }
    __syncthreads();

    const int wave = tid >> 5;           // wave32: channel tile 0..3
    const int lane = tid & 31;
    const int half = lane >> 4;          // lane group 0/1
    const int nl   = lane & 15;

    // A operand (weights, 16x32 bf16): lane row M = lane&15;
    //   lanes 0-15 carry K {k0..k0+7, k0+16..k0+23}, lanes 16-31 are +8.
    const unsigned short* WhRow = Wh + (wave * 16 + nl) * KDIM;
    const unsigned short* WlRow = Wl + (wave * 16 + nl) * KDIM;
    // B operand (patches, 32x16 bf16): lane col N = lane&15;
    //   lanes 0-15 carry K {k0..k0+7, k0+16..k0+23}, lanes 16-31 are +8.
    const unsigned short* PhRow = Ph + nl * KDIM;
    const unsigned short* PlRow = Pl + nl * KDIM;

    union FragU { uint4 q[2]; v16bf v; };

    v8f acc = {};
#pragma unroll
    for (int k0 = 0; k0 < KDIM; k0 += 32) {
        const int aoff = k0 + (half << 3);    // +0 / +8
        const int boff = k0 + (half << 3);    // +0 / +8
        FragU ah, al, bh, bl;
        ah.q[0] = *reinterpret_cast<const uint4*>(WhRow + aoff);
        ah.q[1] = *reinterpret_cast<const uint4*>(WhRow + aoff + 16);
        al.q[0] = *reinterpret_cast<const uint4*>(WlRow + aoff);
        al.q[1] = *reinterpret_cast<const uint4*>(WlRow + aoff + 16);
        bh.q[0] = *reinterpret_cast<const uint4*>(PhRow + boff);
        bh.q[1] = *reinterpret_cast<const uint4*>(PhRow + boff + 16);
        bl.q[0] = *reinterpret_cast<const uint4*>(PlRow + boff);
        bl.q[1] = *reinterpret_cast<const uint4*>(PlRow + boff + 16);

        acc = __builtin_amdgcn_wmma_f32_16x16x32_bf16(false, ah.v, false, bh.v,
                                                      (short)0, acc, false, false);
        acc = __builtin_amdgcn_wmma_f32_16x16x32_bf16(false, ah.v, false, bl.v,
                                                      (short)0, acc, false, false);
        acc = __builtin_amdgcn_wmma_f32_16x16x32_bf16(false, al.v, false, bh.v,
                                                      (short)0, acc, false, false);
    }

    // ---- store: C/D layout: N(pixel)=lane&15, M(channel)=v + 8*half ----
    const int j = j0 + nl;
    if (j < WO) {
#pragma unroll
        for (int v = 0; v < 8; ++v) {
            const int ch = wave * 16 + half * 8 + v;
            out[((b * N_OUT + ch) * HO + i) * WO + j] = acc[v] + bias[ch];
        }
    }
}

// ---------------------------------------------------------------------------
extern "C" void kernel_launch(void* const* d_in, const int* in_sizes, int n_in,
                              void* d_out, int out_size, void* d_ws, size_t ws_size,
                              hipStream_t stream) {
    const float* x        = (const float*)d_in[0];
    const float* weight   = (const float*)d_in[1];
    const float* bias     = (const float*)d_in[2];
    const int*   conn_in  = (const int*)d_in[3];
    const int*   conn_out = (const int*)d_in[4];

    unsigned short* Wh = (unsigned short*)d_ws;                 // 64*576 u16
    unsigned short* Wl = Wh + N_OUT * KDIM;                     // 64*576 u16

    prep_weights<<<dim3(N_OUT), dim3(128), 0, stream>>>(weight, conn_in,
                                                        conn_out, Wh, Wl);

    dim3 grid((WO + 15) / 16, HO, NB);   // 12 x 190 x 4 tiles
    conv_wmma<<<grid, dim3(128), 0, stream>>>(x, bias, Wh, Wl, (float*)d_out);
}
